// GNN_14559939133444
// MI455X (gfx1250) — matrix operand order repaired and checked by reading, not compiled, not run on hardware
//
#include <hip/hip_runtime.h>
#include <math.h>

// CDNA5 / gfx1250: wave32, WMMA bf16 16x16x32 (f32 accum).
// One workgroup (256 thr = 8 wave32) per sample; all per-sample state in LDS.
// Weights pre-swizzled once into WMMA B-fragment order (bf16) in d_ws (L2-resident).
// All three matmul-shaped stages run on WMMA:
//   G = X@X^T (distances), Y = Ahat@X (aggregation), Z = relu(Y@W + b).

typedef __attribute__((ext_vector_type(16))) __bf16 v16bf;
typedef __attribute__((ext_vector_type(8)))  float  v8f;

#define PP   20   // points per sample
#define KNN  8
#define NT   256  // threads per block
#define LDX  512  // fixed row stride of LDS feature buffer (elements)

__device__ __forceinline__ unsigned short f2bf(float f) {
  union { float f; unsigned u; } x; x.f = f;
  unsigned r = x.u + 0x7FFFu + ((x.u >> 16) & 1u);   // round-to-nearest-even
  return (unsigned short)(r >> 16);
}
__device__ __forceinline__ float bf2f(unsigned short s) {
  union { unsigned u; float f; } x; x.u = ((unsigned)s) << 16;
  return x.f;
}

// ---------------------------------------------------------------------------
// Prep: rewrite W1/W2/W3 (fp32, Din x Dout row-major) into bf16 WMMA B-fragment
// order. Fragment f = nt*ktiles + kt is 512 elems: j = lane*16 + i with
//   n = nt*16 + (lane&15),  k = kt*32 + (lane>>4)*16 + i.
// Layer offsets (elems): L1 @0 (64 frags), L2 @64*512 (256), L3 @320*512 (128).
// ---------------------------------------------------------------------------
__global__ __launch_bounds__(256)
void gnn_prep(const float* __restrict__ W1, const float* __restrict__ W2,
              const float* __restrict__ W3, unsigned short* __restrict__ wfrag)
{
  int tid = blockIdx.x * blockDim.x + threadIdx.x;
  if (tid >= 448 * 512) return;
  int Din, Dout, base; const float* W;
  if (tid < 64 * 512)       { Din = 128; Dout = 256; base = 0;         W = W1; }
  else if (tid < 320 * 512) { Din = 256; Dout = 512; base = 64 * 512;  W = W2; }
  else                      { Din = 512; Dout = 128; base = 320 * 512; W = W3; }
  int local = tid - base;
  int f = local >> 9, j = local & 511;
  int lanej = j >> 4, i = j & 15;
  int ktiles = Din >> 5;
  int kt = f % ktiles, nt = f / ktiles;
  int n = nt * 16 + (lanej & 15);
  int k = kt * 32 + (lanej >> 4) * 16 + i;
  wfrag[tid] = f2bf(W[k * Dout + n]);
}

// ---------------------------------------------------------------------------
// Main kernel.
// ---------------------------------------------------------------------------
__global__ __launch_bounds__(256)
void gnn_main(const float* __restrict__ x,
              const float* __restrict__ b1, const float* __restrict__ b2,
              const float* __restrict__ b3,
              const float* __restrict__ W4, const float* __restrict__ b4,
              const float* __restrict__ W5, const float* __restrict__ b5,
              const unsigned short* __restrict__ wfrag,
              float* __restrict__ out)
{
  __shared__ __align__(16) unsigned short sX[32 * LDX];   // 32 KB bf16 features (M padded to 32)
  __shared__ __align__(16) unsigned short sAdjB[32 * 32]; // Ahat, bf16, zero-padded
  __shared__ float sGram[32 * 32];   // padded: D-fragments stored unconditionally
  __shared__ float sD2[PP * PP];
  __shared__ int   sNbr[PP * KNN];
  __shared__ int   sDeg[PP];
  __shared__ float sDinv[PP];
  __shared__ float sH[128];
  __shared__ float sG[64];
  __shared__ float sL[3];

  const int tid  = threadIdx.x;
  const int lane = tid & 31;
  const int wv   = tid >> 5;
  const int half = lane >> 4;
  const int ln15 = lane & 15;
  const int samp = blockIdx.x;

  // ---- load sample (20x128 fp32 -> bf16); zero padding rows 20..31 ----
  // Zeroing is mandatory: WMMA propagates NaN even through 0*x, and LDS is
  // uninitialized at launch.
  const float* xs = x + (size_t)samp * (PP * 128);
  for (int e = tid; e < 32 * 128; e += NT) {
    int r = e >> 7, c = e & 127;
    sX[r * LDX + c] = (r < PP) ? f2bf(xs[r * 128 + c]) : (unsigned short)0;
  }
  __syncthreads();

  const int dins[3]  = {128, 256, 512};
  const int douts[3] = {256, 512, 128};
  const int foff[3]  = {0, 64 * 512, 320 * 512};
  const float* biases[3] = {b1, b2, b3};

#pragma unroll
  for (int layer = 0; layer < 3; ++layer) {
    const int Din    = dins[layer];
    const int ktiles = Din >> 5;
    const int ntiles = douts[layer] >> 4;

    // ================= Phase A: Gram G = X @ X^T via WMMA ==================
    // 4 output tiles (2x2 over 32x32), one per wave 0..3. Padding region of
    // sGram receives finite garbage (rows 20..31 of sX are always finite) and
    // is never read.
    if (wv < 4) {
      int mt = wv & 1, nt2 = wv >> 1;
      v8f g = {0.f, 0.f, 0.f, 0.f, 0.f, 0.f, 0.f, 0.f};
      for (int kt = 0; kt < ktiles; ++kt) {
        // A fragment: rows mt*16.., K = kt*32.., pairwise 32-bit LDS loads
        union { v16bf v; unsigned u[8]; } af;
        const unsigned* ap =
            (const unsigned*)&sX[(mt * 16 + ln15) * LDX + kt * 32];
#pragma unroll
        for (int pr = 0; pr < 8; ++pr) {
          int kb = half * 16 + 2 * pr;
          af.u[pr] = ap[kb >> 1];
        }
        // B fragment = X^T: B[k][n] = X[n][k] -> per lane 32 contiguous bytes
        // of row n = nt2*16 + (lane&15): two ds_load_b128.
        union { v16bf v; uint4 q[2]; } bf;
        const uint4* bp = (const uint4*)
            &sX[(nt2 * 16 + ln15) * LDX + kt * 32 + half * 16];
        bf.q[0] = bp[0];
        bf.q[1] = bp[1];
        g = __builtin_amdgcn_wmma_f32_16x16x32_bf16(false, af.v, false, bf.v,
                                                    (short)0, g, false, false);
      }
      // unconditional, branch-free D store into padded 32x32 buffer
      int col = nt2 * 16 + ln15;
#pragma unroll
      for (int r = 0; r < 8; ++r) {
        int row = mt * 16 + half * 8 + r;
        sGram[row * 32 + col] = g[r];
      }
    }
    __syncthreads();

    // ---- d2[i][j] = G[ii] + G[jj] - 2 G[ij]  (self -> +inf) ----
    for (int p = tid; p < PP * PP; p += NT) {
      int i = p / PP, j = p % PP;
      sD2[p] = (i == j) ? 3.0e38f
                        : sGram[i * 32 + i] + sGram[j * 32 + j]
                          - 2.f * sGram[i * 32 + j];
    }
    __syncthreads();

    // ---- K-NN selection per row (stable: strict <, ascending scan) ----
    if (tid < PP) {
      sDeg[tid] = 1;                               // self loop
      float* row = &sD2[tid * PP];
      for (int k = 0; k < KNN; ++k) {
        float best = 3.9e38f; int bj = 0;
        for (int j = 0; j < PP; ++j) { float v = row[j]; if (v < best) { best = v; bj = j; } }
        row[bj] = 3.9e38f;
        sNbr[tid * KNN + k] = bj;
      }
    }
    __syncthreads();
    if (tid < PP)
      for (int k = 0; k < KNN; ++k) atomicAdd(&sDeg[sNbr[tid * KNN + k]], 1);
    __syncthreads();
    if (tid < PP) sDinv[tid] = rsqrtf((float)sDeg[tid]);
    __syncthreads();

    // ---- Ahat (32x32 bf16, zero-padded): Ahat[t][s]=dinv[s]dinv[t] ----
    for (int p = tid; p < 512; p += NT) ((unsigned*)sAdjB)[p] = 0u;
    __syncthreads();
    if (tid < PP) {
      int i = tid;
      sAdjB[i * 32 + i] = f2bf(sDinv[i] * sDinv[i]);
      for (int k = 0; k < KNN; ++k) {
        int j = sNbr[i * KNN + k];
        sAdjB[j * 32 + i] = f2bf(sDinv[i] * sDinv[j]);
      }
    }
    __syncthreads();

    // ================= Phase C: Y = Ahat @ X via WMMA ======================
    // M=32 (Ahat rows), K=32 (points, zero-padded), N=Din. One k-tile.
    // Ahat rows >=20 are zero -> D-store re-zeroes padding rows of sX.
    {
      const int ntilesY = Din >> 4;            // 8 / 16 / 32
      const int tpwY = (2 * ntilesY) >> 3;     // 2 / 4 / 8 tiles per wave
      v8f acc[8];
#pragma unroll
      for (int t = 0; t < tpwY; ++t) {
        int tile = wv + (t << 3);
        int mt = tile & 1, ntY = tile >> 1;
        // A fragment from Ahat (row stride 32, pairwise 32-bit loads)
        union { v16bf v; unsigned u[8]; } af;
        const unsigned* ap = (const unsigned*)&sAdjB[(mt * 16 + ln15) * 32];
#pragma unroll
        for (int pr = 0; pr < 8; ++pr) {
          int kb = half * 16 + 2 * pr;
          af.u[pr] = ap[kb >> 1];
        }
        // B fragment = X: B[k][n] -> strided 16-bit LDS loads (k varies)
        union { v16bf v; unsigned short s[16]; } bf;
        int colY = ntY * 16 + ln15;
#pragma unroll
        for (int i = 0; i < 16; ++i)
          bf.s[i] = sX[(half * 16 + i) * LDX + colY];
        v8f c0 = {0.f, 0.f, 0.f, 0.f, 0.f, 0.f, 0.f, 0.f};
        acc[t] = __builtin_amdgcn_wmma_f32_16x16x32_bf16(false, af.v, false, bf.v,
                                                         (short)0, c0, false, false);
      }
      __syncthreads();                         // all reads of sX done
#pragma unroll
      for (int t = 0; t < tpwY; ++t) {
        int tile = wv + (t << 3);
        int mt = tile & 1, ntY = tile >> 1;
        int colY = ntY * 16 + ln15;
#pragma unroll
        for (int r = 0; r < 8; ++r) {
          int row = mt * 16 + half * 8 + r;
          sX[row * LDX + colY] = f2bf(acc[t][r]);
        }
      }
      __syncthreads();
    }

    // ================= Phase D: Z = relu(Y @ W + b) via WMMA ===============
    {
      const unsigned short* wl = wfrag + foff[layer];
      const float* bias = biases[layer];
      const int tpw = (2 * ntiles) >> 3;       // 4 / 8 / 2 tiles per wave
      v8f acc[8];
#pragma unroll
      for (int t = 0; t < tpw; ++t) {
        int tile = wv + (t << 3);
        int mt = tile & 1, nt = tile >> 1;
        v8f c = {0.f, 0.f, 0.f, 0.f, 0.f, 0.f, 0.f, 0.f};
        for (int kt = 0; kt < ktiles; ++kt) {
          union { v16bf v; unsigned u[8]; } af;
          const unsigned* ap =
              (const unsigned*)&sX[(mt * 16 + ln15) * LDX + kt * 32];
#pragma unroll
          for (int pr = 0; pr < 8; ++pr) {
            int kb = half * 16 + 2 * pr;
            af.u[pr] = ap[kb >> 1];
          }
          union { v16bf v; uint4 q[2]; } bf;
          const uint4* bp =
              (const uint4*)(wl + (size_t)(nt * ktiles + kt) * 512) + lane * 2;
          bf.q[0] = bp[0];
          bf.q[1] = bp[1];
          if (kt + 1 < ktiles)
            __builtin_prefetch(wl + (size_t)(nt * ktiles + kt + 1) * 512 + lane * 16, 0, 1);
          c = __builtin_amdgcn_wmma_f32_16x16x32_bf16(false, af.v, false, bf.v,
                                                      (short)0, c, false, false);
        }
        acc[t] = c;
      }
      __syncthreads();                         // all WMMA reads of sX done
#pragma unroll
      for (int t = 0; t < tpw; ++t) {
        int tile = wv + (t << 3);
        int mt = tile & 1, nt = tile >> 1;
        int col = nt * 16 + ln15;
        float bv = bias[col];
#pragma unroll
        for (int r = 0; r < 8; ++r) {
          int row = mt * 16 + half * 8 + r;
          float z = acc[t][r] + bv;
          sX[row * LDX + col] = f2bf(z > 0.f ? z : 0.f);
        }
      }
      __syncthreads();
    }
  }

  // ---- mean pool over 20 rows (128 feats) ----
  if (tid < 128) {
    float s = 0.f;
    for (int r = 0; r < PP; ++r) s += bf2f(sX[r * LDX + tid]);
    sH[tid] = s * (1.0f / PP);
  }
  __syncthreads();
  // ---- FC 128->64 + relu ----
  if (tid < 64) {
    float s = b4[tid];
    for (int c = 0; c < 128; ++c) s = fmaf(sH[c], W4[c * 64 + tid], s);
    sG[tid] = s > 0.f ? s : 0.f;
  }
  __syncthreads();
  // ---- FC 64->3 ----
  if (tid < 3) {
    float s = b5[tid];
    for (int c = 0; c < 64; ++c) s = fmaf(sG[c], W5[c * 3 + tid], s);
    sL[tid] = s;
  }
  __syncthreads();
  // ---- softmax ----
  if (tid < 3) {
    float m  = fmaxf(sL[0], fmaxf(sL[1], sL[2]));
    float e0 = __expf(sL[0] - m), e1 = __expf(sL[1] - m), e2 = __expf(sL[2] - m);
    float inv = 1.0f / (e0 + e1 + e2);
    float e = (tid == 0) ? e0 : (tid == 1) ? e1 : e2;
    out[samp * 3 + tid] = e * inv;
  }
}

extern "C" void kernel_launch(void* const* d_in, const int* in_sizes, int n_in,
                              void* d_out, int out_size, void* d_ws, size_t ws_size,
                              hipStream_t stream) {
  const float* x  = (const float*)d_in[0];
  const float* W1 = (const float*)d_in[1];
  const float* b1 = (const float*)d_in[2];
  const float* W2 = (const float*)d_in[3];
  const float* b2 = (const float*)d_in[4];
  const float* W3 = (const float*)d_in[5];
  const float* b3 = (const float*)d_in[6];
  const float* W4 = (const float*)d_in[7];
  const float* b4 = (const float*)d_in[8];
  const float* W5 = (const float*)d_in[9];
  const float* b5 = (const float*)d_in[10];

  unsigned short* wfrag = (unsigned short*)d_ws;   // needs 448*512*2 = 458752 B

  const int N = in_sizes[0] / (PP * 128);          // 8192 samples

  gnn_prep<<<(448 * 512 + 255) / 256, 256, 0, stream>>>(W1, W2, W3, wfrag);
  gnn_main<<<N, 256, 0, stream>>>(x, b1, b2, b3, W4, b4, W5, b5, wfrag,
                                  (float*)d_out);
}
